// Generator_83854941487754
// MI455X (gfx1250) — matrix operand order repaired and checked by reading, not compiled, not run on hardware
//
#include <hip/hip_runtime.h>

// ---------------------------------------------------------------------------
// Neural SDE (reversible Heun) fully fused for MI455X (gfx1250, wave32, WMMA).
//   - bf16 weights (converted once per launch into d_ws), f32 accumulation
//   - v_wmma_f32_16x16x32_bf16 for every GEMM; QUAD-tile interleaving so the
//     bf16 WMMA->WMMA RAW hazard window (1 NOP + 4 coexec) is fully hidden
//   - diffusion output contracted with dW in-register (never materialized)
//   - readout fused per step (trajectory never stored)
// ---------------------------------------------------------------------------

typedef __bf16 v16bf __attribute__((ext_vector_type(16)));
typedef float  v8f   __attribute__((ext_vector_type(8)));
typedef unsigned int uu4 __attribute__((ext_vector_type(4)));

// bf16 weight workspace layout (offsets in ushort elements)
#define OFF_EW0   0        // 64x256
#define OFF_EW1   16384    // 256x128
#define OFF_DW0   49152    // 129x256 (row 0 = t-row)
#define OFF_DW1   82176    // 256x256
#define OFF_DW2   147712   // 256x128
#define OFF_GW0   180480   // 129x256
#define OFF_GW1   213504   // 256x256
#define OFF_GW2   279040   // 256x2048
#define OFF_RW0   803328   // 129x256
#define OFF_RW1P  836352   // 256x16 (8 real cols, zero padded)

__device__ __forceinline__ unsigned short f2bf(float f) {
  // round-half-up: single VALU add feeding ds_store_b16_d16_hi
  return (unsigned short)((__float_as_uint(f) + 0x8000u) >> 16);
}
__device__ __forceinline__ float bf2f(unsigned short s) {
  return __uint_as_float(((unsigned)s) << 16);
}
__device__ __forceinline__ float relu1(float x) {
  // v_med3_f32: single-instruction ReLU (median of {x, 0, +huge})
  return __builtin_amdgcn_fmed3f(x, 0.0f, 3.0e38f);
}

__device__ __forceinline__ v8f bcast8(float x) {
  v8f a;
#pragma unroll
  for (int i = 0; i < 8; ++i) a[i] = x;
  return a;
}

// 16-lane (N-group) sum reduction; masks 1/2/4/8 stay inside each half-wave.
__device__ __forceinline__ float red16(float v) {
  v += __shfl_xor(v, 1, 32);
  v += __shfl_xor(v, 2, 32);
  v += __shfl_xor(v, 4, 32);
  v += __shfl_xor(v, 8, 32);
  return v;
}

__device__ __forceinline__ void load_afrag(const unsigned short* __restrict__ A, int lda,
                                           int k0, int lane, v16bf& av) {
  const int m = lane & 15, hi = lane >> 4;
  const unsigned short* pa = A + m * lda + k0 + hi * 8;
  union { uu4 u[2]; v16bf v; } a;
  a.u[0] = *(const uu4*)(pa);        // K = base+0..7
  a.u[1] = *(const uu4*)(pa + 16);   // K = base+16..23
  av = a.v;
}

// Quad-tile accumulate: columns [n0, n0+64) as four 16-wide tiles.
// One shared A-fragment feeds 4 independent WMMA chains: dependent WMMAs are
// always >=3 instructions apart, fully covering the bf16 hazard window, and
// A-fragment LDS reads are amortized 4x. B reads: per k row, the 4 tiles cover 128 contiguous bytes.
__device__ __forceinline__ void wmma_accum4(const unsigned short* __restrict__ A, int lda,
                                            const unsigned short* __restrict__ W, int ldw,
                                            int n0, int K, v8f (&acc)[4], int lane) {
  for (int k0 = 0; k0 < K; k0 += 32) {
    v16bf av;
    load_afrag(A, lda, k0, lane, av);
    const unsigned short* pb = W + (size_t)(k0 + 8 * (lane >> 4)) * ldw + n0 + (lane & 15);
#pragma unroll
    for (int q = 0; q < 4; ++q) {
      union { unsigned short s[16]; v16bf v; } b;
#pragma unroll
      for (int s = 0; s < 16; ++s)
        b.s[s] = pb[(size_t)(s < 8 ? s : s + 8) * ldw + 16 * q];
      acc[q] = __builtin_amdgcn_wmma_f32_16x16x32_bf16(false, av, false, b.v,
                                                       (short)0, acc[q], false, false);
    }
  }
}

// Single 16-col tile, K split into two interleaved chains (readout tail).
__device__ __forceinline__ v8f wmma_accum_split(const unsigned short* __restrict__ A, int lda,
                                                const unsigned short* __restrict__ W, int ldw,
                                                int n0, int K, v8f acc, int lane) {
  v8f acc1 = bcast8(0.0f);
  for (int k0 = 0; k0 < K; k0 += 64) {
    v16bf a0, a1;
    load_afrag(A, lda, k0, lane, a0);
    load_afrag(A, lda, k0 + 32, lane, a1);
    const unsigned short* pb0 = W + (size_t)(k0 + 8 * (lane >> 4)) * ldw + n0 + (lane & 15);
    const unsigned short* pb1 = W + (size_t)(k0 + 32 + 8 * (lane >> 4)) * ldw + n0 + (lane & 15);
    union { unsigned short s[16]; v16bf v; } b0, b1;
#pragma unroll
    for (int s = 0; s < 16; ++s) {
      b0.s[s] = pb0[(size_t)(s < 8 ? s : s + 8) * ldw];
      b1.s[s] = pb1[(size_t)(s < 8 ? s : s + 8) * ldw];
    }
    acc  = __builtin_amdgcn_wmma_f32_16x16x32_bf16(false, a0, false, b0.v,
                                                   (short)0, acc,  false, false);
    acc1 = __builtin_amdgcn_wmma_f32_16x16x32_bf16(false, a1, false, b1.v,
                                                   (short)0, acc1, false, false);
  }
#pragma unroll
  for (int i = 0; i < 8; ++i) acc[i] += acc1[i];
  return acc;
}

// One MLP hidden layer: out = relu(A @ W + bias (+ t * trow)), bf16 to LDS.
// N must be a multiple of 64 (tiles processed in quads).
__device__ __forceinline__ void layer_relu(const unsigned short* __restrict__ A, int lda, int K,
                                           const unsigned short* __restrict__ W, int ldw, int N,
                                           const float* __restrict__ bias,
                                           const unsigned short* __restrict__ trow, float t,
                                           unsigned short* __restrict__ out, int ldo, int lane) {
  const int nlo = lane & 15, hi = lane >> 4;
  for (int n0 = 0; n0 < N; n0 += 64) {
    v8f acc[4];
#pragma unroll
    for (int q = 0; q < 4; ++q) {
      int c = n0 + 16 * q + nlo;
      float b = bias[c];
      if (trow) b += t * bf2f(trow[c]);
      acc[q] = bcast8(b);
    }
    wmma_accum4(A, lda, W, ldw, n0, K, acc, lane);
#pragma unroll
    for (int q = 0; q < 4; ++q) {
      int c = n0 + 16 * q + nlo;
#pragma unroll
      for (int i = 0; i < 8; ++i)
        out[(i + 8 * hi) * ldo + c] = f2bf(relu1(acc[q][i]));
    }
  }
}

// ---------------------------------------------------------------------------
// weight conversion kernels (run every launch; deterministic, graph-safe)
// ---------------------------------------------------------------------------
__global__ void cvt_bf16(const float* __restrict__ s, unsigned short* __restrict__ d, int n) {
  int i = blockIdx.x * blockDim.x + threadIdx.x;
  if (i < n) d[i] = f2bf(s[i]);
}
__global__ void cvt_pad_readout(const float* __restrict__ s, unsigned short* __restrict__ d) {
  int i = blockIdx.x * blockDim.x + threadIdx.x;
  if (i < 256 * 16) {
    int k = i >> 4, n = i & 15;
    d[i] = (n < 8) ? f2bf(s[k * 8 + n]) : (unsigned short)0;
  }
}

// ---------------------------------------------------------------------------
// fully fused SDE integrator + per-step readout
// grid = 128 blocks x 64 threads (2 waves); each wave owns 16 batch rows.
// ---------------------------------------------------------------------------
__global__ __launch_bounds__(64) void sde_fused(
    const float* __restrict__ noise, const float* __restrict__ dWp,
    const unsigned short* __restrict__ wb,
    const float* __restrict__ eb0, const float* __restrict__ eb1,
    const float* __restrict__ db0, const float* __restrict__ db1, const float* __restrict__ db2,
    const float* __restrict__ gb0, const float* __restrict__ gb1, const float* __restrict__ gb2,
    const float* __restrict__ rb0, const float* __restrict__ rb1,
    float* __restrict__ out) {
  __shared__ __align__(16) unsigned short sX[2][16 * 128];   // x / z1 / y  (bf16)
  __shared__ __align__(16) unsigned short sH1[2][16 * 256];  // hidden 1    (bf16)
  __shared__ __align__(16) unsigned short sH2[2][16 * 256];  // hidden 2    (bf16)

  const int w = threadIdx.x >> 5, lane = threadIdx.x & 31;
  const int nlo = lane & 15, hi = lane >> 4;
  const int rb = blockIdx.x * 32 + w * 16;   // first batch row of this wave

  unsigned short* X  = sX[w];
  unsigned short* H1 = sH1[w];
  unsigned short* H2 = sH2[w];

  const unsigned short* EW0  = wb + OFF_EW0;
  const unsigned short* EW1  = wb + OFF_EW1;
  const unsigned short* DW0  = wb + OFF_DW0;
  const unsigned short* DW1  = wb + OFF_DW1;
  const unsigned short* DW2  = wb + OFF_DW2;
  const unsigned short* GW0  = wb + OFF_GW0;
  const unsigned short* GW1  = wb + OFF_GW1;
  const unsigned short* GW2  = wb + OFF_GW2;
  const unsigned short* RW0  = wb + OFF_RW0;
  const unsigned short* RW1P = wb + OFF_RW1P;

  // per-lane SDE state in C-fragment layout: element (m = i+8*hi, col = 16j+nlo)
  float y[64], z[64], f[64], gdw[64];

  // -------- embed: x0 = MLP(init_noise) --------
  for (int idx = lane; idx < 16 * 64; idx += 32) {
    int r = idx >> 6, c = idx & 63;
    X[idx] = f2bf(noise[(size_t)(rb + r) * 64 + c]);  // 16x64 staging, stride 64
  }
  layer_relu(X, 64, 64, EW0, 256, 256, eb0, nullptr, 0.f, H1, 256, lane);
#pragma unroll
  for (int j = 0; j < 8; j += 4) {
    v8f acc[4];
#pragma unroll
    for (int q = 0; q < 4; ++q) acc[q] = bcast8(eb1[16 * (j + q) + nlo]);
    wmma_accum4(H1, 256, EW1, 128, 16 * j, 256, acc, lane);
#pragma unroll
    for (int q = 0; q < 4; ++q)
#pragma unroll
      for (int i = 0; i < 8; ++i) {
        y[(j + q) * 8 + i] = acc[q][i];
        z[(j + q) * 8 + i] = acc[q][i];
        X[(i + 8 * hi) * 128 + 16 * (j + q) + nlo] = f2bf(acc[q][i]);
      }
  }

  // -------- f0 = drift(0, x0) --------
  layer_relu(X, 128, 128, DW0 + 256, 256, 256, db0, DW0, 0.f, H1, 256, lane);
  layer_relu(H1, 256, 256, DW1, 256, 256, db1, nullptr, 0.f, H2, 256, lane);
#pragma unroll
  for (int j = 0; j < 8; j += 4) {
    v8f acc[4];
#pragma unroll
    for (int q = 0; q < 4; ++q) acc[q] = bcast8(db2[16 * (j + q) + nlo]);
    wmma_accum4(H2, 256, DW2, 128, 16 * j, 256, acc, lane);
#pragma unroll
    for (int q = 0; q < 4; ++q)
#pragma unroll
      for (int i = 0; i < 8; ++i) f[(j + q) * 8 + i] = acc[q][i];
  }

  // -------- gdw = g(0, x0) . dW[0] (diffusion contracted on the fly) --------
  layer_relu(X, 128, 128, GW0 + 256, 256, 256, gb0, GW0, 0.f, H1, 256, lane);
  layer_relu(H1, 256, 256, GW1, 256, 256, gb1, nullptr, 0.f, H2, 256, lane);
  float dwn[8];
#pragma unroll
  for (int i = 0; i < 8; ++i)
    dwn[i] = dWp[((size_t)(rb + i + 8 * hi)) * 16 + nlo];  // dW[0][row][n]
#pragma unroll
  for (int j = 0; j < 8; ++j) {
#pragma unroll 1
    for (int hh = 0; hh < 16; hh += 4) {
      int h = j * 16 + hh;                       // hidden units h..h+3
      v8f acc[4];
#pragma unroll
      for (int q = 0; q < 4; ++q) acc[q] = bcast8(gb2[16 * (h + q) + nlo]);
      wmma_accum4(H2, 256, GW2, 2048, 16 * h, 256, acc, lane);
#pragma unroll
      for (int q = 0; q < 4; ++q)
#pragma unroll
        for (int i = 0; i < 8; ++i) {
          float p = red16(acc[q][i] * dwn[i]);   // sum over 16 noise dims
          if (nlo == hh + q) gdw[j * 8 + i] = p; // lands in C-fragment layout
        }
    }
  }

  // -------- fused readout of x_t (X holds x_t bf16) --------
  auto readout = [&](int t) {
    layer_relu(X, 128, 128, RW0 + 256, 256, 256, rb0, RW0, (float)t, H1, 256, lane);
    v8f acc = bcast8((nlo < 8) ? rb1[nlo] : 0.f);
    acc = wmma_accum_split(H1, 256, RW1P, 16, 0, 256, acc, lane);
    if (nlo < 8) {
#pragma unroll
      for (int i = 0; i < 8; ++i)
        out[(((size_t)(rb + i + 8 * hi)) * 64 + t) * 8 + nlo] = acc[i];
    }
  };
  readout(0);

  // -------- reversible Heun steps --------
#pragma unroll 1
  for (int t = 1; t < 64; ++t) {
    float tf = (float)t;
    float dwv[8];
#pragma unroll
    for (int i = 0; i < 8; ++i) dwv[i] = dwn[i];           // dW[t-1]
    if (t < 63) {
#pragma unroll
      for (int i = 0; i < 8; ++i)
        dwn[i] = dWp[((size_t)t * 4096 + rb + i + 8 * hi) * 16 + nlo];  // dW[t]
    } else {
#pragma unroll
      for (int i = 0; i < 8; ++i) dwn[i] = 0.f;
    }

    // z1 = 2y - z + f*dt + g0.dw ; pre-accumulate f0/g0 halves into y
#pragma unroll
    for (int j = 0; j < 8; ++j)
#pragma unroll
      for (int i = 0; i < 8; ++i) {
        int idx = j * 8 + i;
        float zn = 2.f * y[idx] - z[idx] + f[idx] + gdw[idx];
        y[idx] += 0.5f * (f[idx] + gdw[idx]);
        z[idx] = zn;
        X[(i + 8 * hi) * 128 + 16 * j + nlo] = f2bf(zn);
      }

    // drift: f1 ; y += 0.5*f1
    layer_relu(X, 128, 128, DW0 + 256, 256, 256, db0, DW0, tf, H1, 256, lane);
    layer_relu(H1, 256, 256, DW1, 256, 256, db1, nullptr, 0.f, H2, 256, lane);
#pragma unroll
    for (int j = 0; j < 8; j += 4) {
      v8f acc[4];
#pragma unroll
      for (int q = 0; q < 4; ++q) acc[q] = bcast8(db2[16 * (j + q) + nlo]);
      wmma_accum4(H2, 256, DW2, 128, 16 * j, 256, acc, lane);
#pragma unroll
      for (int q = 0; q < 4; ++q)
#pragma unroll
        for (int i = 0; i < 8; ++i) {
          f[(j + q) * 8 + i] = acc[q][i];
          y[(j + q) * 8 + i] += 0.5f * acc[q][i];
        }
    }

    // diffusion: contract each 16x16 tile with dW[t-1] (-> y) and dW[t] (-> gdw)
    layer_relu(X, 128, 128, GW0 + 256, 256, 256, gb0, GW0, tf, H1, 256, lane);
    layer_relu(H1, 256, 256, GW1, 256, 256, gb1, nullptr, 0.f, H2, 256, lane);
#pragma unroll
    for (int j = 0; j < 8; ++j) {
#pragma unroll 1
      for (int hh = 0; hh < 16; hh += 4) {
        int h = j * 16 + hh;
        v8f acc[4];
#pragma unroll
        for (int q = 0; q < 4; ++q) acc[q] = bcast8(gb2[16 * (h + q) + nlo]);
        wmma_accum4(H2, 256, GW2, 2048, 16 * h, 256, acc, lane);
#pragma unroll
        for (int q = 0; q < 4; ++q)
#pragma unroll
          for (int i = 0; i < 8; ++i) {
            float g = acc[q][i];
            float pv = red16(g * dwv[i]);
            float pn = red16(g * dwn[i]);
            if (nlo == hh + q) {
              y[j * 8 + i] += 0.5f * pv;   // 0.5*(g1 . dW[t-1])
              gdw[j * 8 + i] = pn;         // g1 . dW[t]  (next step's g0.dw)
            }
          }
      }
    }

    // publish y_t and run readout
#pragma unroll
    for (int j = 0; j < 8; ++j)
#pragma unroll
      for (int i = 0; i < 8; ++i)
        X[(i + 8 * hi) * 128 + 16 * j + nlo] = f2bf(y[j * 8 + i]);
    readout(t);
  }
}

// ---------------------------------------------------------------------------
extern "C" void kernel_launch(void* const* d_in, const int* in_sizes, int n_in,
                              void* d_out, int out_size, void* d_ws, size_t ws_size,
                              hipStream_t stream) {
  (void)in_sizes; (void)n_in; (void)out_size; (void)ws_size;
  const float* noise = (const float*)d_in[0];
  const float* dWp   = (const float*)d_in[1];
  unsigned short* wb = (unsigned short*)d_ws;   // ~1.7 MB of bf16 weights

  auto C = [&](int argi, int off, int n) {
    cvt_bf16<<<(n + 255) / 256, 256, 0, stream>>>((const float*)d_in[argi], wb + off, n);
  };
  C(2,  OFF_EW0, 64 * 256);
  C(4,  OFF_EW1, 256 * 128);
  C(6,  OFF_DW0, 129 * 256);
  C(8,  OFF_DW1, 256 * 256);
  C(10, OFF_DW2, 256 * 128);
  C(12, OFF_GW0, 129 * 256);
  C(14, OFF_GW1, 256 * 256);
  C(16, OFF_GW2, 256 * 2048);
  C(18, OFF_RW0, 129 * 256);
  cvt_pad_readout<<<(256 * 16 + 255) / 256, 256, 0, stream>>>(
      (const float*)d_in[20], wb + OFF_RW1P);

  sde_fused<<<128, 64, 0, stream>>>(
      noise, dWp, wb,
      (const float*)d_in[3],  (const float*)d_in[5],
      (const float*)d_in[7],  (const float*)d_in[9],  (const float*)d_in[11],
      (const float*)d_in[13], (const float*)d_in[15], (const float*)d_in[17],
      (const float*)d_in[19], (const float*)d_in[21],
      (float*)d_out);
}
